// Model_40441412059277
// MI455X (gfx1250) — matrix operand order repaired and checked by reading, not compiled, not run on hardware
//
#include <hip/hip_runtime.h>
#include <hip/hip_bf16.h>
#include <cstdint>

// ---------------------------------------------------------------------------
// GNN generator forward for gfx1250 (MI455X).
// Dense projections: wave32 WMMA f16 (f32 accumulate). Weights are converted
// to f16 and pre-packed in LDS in the exact WMMA B-fragment layout, so each
// fragment is fetched with one 32-byte LDS vector load (2x ds_load_b128).
// Epilogue is branch-free (immediate-offset stores) for all full 16-row tiles.
// Sparse attention: ordered-uint atomic-max segment softmax + f32 atomic adds.
// ---------------------------------------------------------------------------

typedef __attribute__((ext_vector_type(16))) _Float16 v16h;
typedef __attribute__((ext_vector_type(8)))  float    v8f;

namespace {

constexpr int   kN   = 100000;           // nodes
constexpr int   kE   = 400000;           // edges
constexpr int   kNE  = 400000;           // non-edges
constexpr int   kG   = 512;              // graphs
constexpr int   kD   = 64;               // embedding dim
constexpr int   kNA  = kN + kG;          // augmented node count
constexpr int   kEA  = kE + 2 * kN;      // augmented edge count
constexpr int   kE2  = kE / 2;
constexpr float kEps = 1e-7f;

__device__ __forceinline__ unsigned f2ord(float f) {
  unsigned u = __float_as_uint(f);
  return (u & 0x80000000u) ? ~u : (u | 0x80000000u);
}
__device__ __forceinline__ float ord2f(unsigned u) {
  u = (u & 0x80000000u) ? (u & 0x7fffffffu) : ~u;
  return __uint_as_float(u);
}

// ---------------- WMMA GEMM: C[M,NOUT] = act(A[M,K] @ W[K,NOUT] + bias) -----
// 256 threads = 8 waves; each wave -> 16 output rows; block -> 128 rows.
// ACT: 0 = identity, 1 = leaky_relu(0.01).
template <int K, int NOUT, int ACT>
__global__ __launch_bounds__(256)
void gemm_wmma(const float* __restrict__ A, const float* __restrict__ W,
               const float* __restrict__ bias, float* __restrict__ C, int M) {
  constexpr int  NT    = NOUT / 16;
  constexpr bool KFULL = (K % 32) == 0;
  constexpr int  KCH   = (K + 31) / 32;
  constexpr int  SLOTS = KCH * NT * 32;           // one 16-half slot per lane
  __shared__ __align__(32) _Float16 sW[SLOTS * 16];

  const int tid = threadIdx.x;
  // ---- pack W -> LDS in B-fragment layout ---------------------------------
  // B fragment (32x16 f16): lanes 0-15: halves 0-7 -> K=k0+0..7, 8-15 -> k0+16..23, N=t*16+lane;
  //                         lanes 16-31: halves 0-7 -> K=k0+8..15, 8-15 -> k0+24..31.
  for (int slot = tid; slot < SLOTS; slot += 256) {
    const int lane  = slot & 31;
    const int t     = (slot >> 5) % NT;
    const int kc    = slot / (32 * NT);
    const int col   = t * 16 + (lane & 15);
    const int kbase = kc * 32 + ((lane >= 16) ? 8 : 0);
    _Float16* dp = &sW[(size_t)slot * 16];
#pragma unroll
    for (int h = 0; h < 16; ++h) {
      const int kk = kbase + (h < 8 ? h : h + 8);
      dp[h] = (KFULL || kk < K) ? (_Float16)W[kk * NOUT + col] : (_Float16)0.f;
    }
  }
  __syncthreads();

  const int  wave = tid >> 5;
  const int  lane = tid & 31;
  const int  ln   = lane & 15;
  const bool hi   = lane >= 16;
  const int  row0 = blockIdx.x * 128 + wave * 16;

  // bias per n-tile column, hoisted out of the epilogue
  float bvs[NT];
#pragma unroll
  for (int t = 0; t < NT; ++t) bvs[t] = bias ? bias[t * 16 + ln] : 0.f;

  v8f acc[NT];
#pragma unroll
  for (int t = 0; t < NT; ++t)
#pragma unroll
    for (int i = 0; i < 8; ++i) acc[t][i] = 0.f;

  for (int kc = 0; kc < KCH; ++kc) {
    const int k0 = kc * 32;
    // A fragment (16x32 f16). ISA layout: lanes 0-15 halves 0-7 -> K=k0+0..7,
    // halves 8-15 -> k0+16..23; lanes 16-31 -> k0+8..15 / k0+24..31.
    v16h a;
    const int arow = row0 + ln;
    const int kA   = k0 + (hi ? 8 : 0);
    if (arow < M) {
      const float* ap = A + (size_t)arow * K;
#pragma unroll
      for (int h = 0; h < 8; ++h) {
        const int kk = kA + h;
        a[h] = (KFULL || kk < K) ? (_Float16)ap[kk] : (_Float16)0.f;
      }
#pragma unroll
      for (int h = 0; h < 8; ++h) {
        const int kk = kA + 16 + h;
        a[8 + h] = (KFULL || kk < K) ? (_Float16)ap[kk] : (_Float16)0.f;
      }
    } else {
#pragma unroll
      for (int h = 0; h < 16; ++h) a[h] = (_Float16)0.f;
    }
#pragma unroll
    for (int t = 0; t < NT; ++t) {
      const v16h b = *(const v16h*)&sW[(size_t)((kc * NT + t) * 32 + lane) * 16];
      acc[t] = __builtin_amdgcn_wmma_f32_16x16x32_f16(false, a, false, b,
                                                      (short)0, acc[t], false, false);
    }
  }

  // C/D layout: lanes 0-15 VGPR i -> M=row0+i, N=lane; lanes 16-31 -> M=row0+8+i.
  const int rbase = row0 + (hi ? 8 : 0);
  if (row0 + 16 <= M) {
    // full tile: branch-free, immediate-offset stores off one base address
#pragma unroll
    for (int t = 0; t < NT; ++t) {
      float* cp = C + (size_t)rbase * NOUT + t * 16 + ln;
#pragma unroll
      for (int i = 0; i < 8; ++i) {
        float v = acc[t][i] + bvs[t];
        if (ACT == 1) v = (v > 0.f) ? v : 0.01f * v;
        cp[(size_t)i * NOUT] = v;
      }
    }
  } else {
#pragma unroll
    for (int t = 0; t < NT; ++t) {
      float* cp = C + (size_t)rbase * NOUT + t * 16 + ln;
#pragma unroll
      for (int i = 0; i < 8; ++i) {
        if (rbase + i < M) {
          float v = acc[t][i] + bvs[t];
          if (ACT == 1) v = (v > 0.f) ? v : 0.01f * v;
          cp[(size_t)i * NOUT] = v;
        }
      }
    }
  }
}

// ---------------- elementwise / utility kernels -----------------------------
__global__ void zero_kernel(float* p, long long n) {
  long long i = (long long)blockIdx.x * blockDim.x + threadIdx.x;
  if (i < n) p[i] = 0.f;
}
__global__ void copy_kernel(float* d, const float* s, long long n) {
  long long i = (long long)blockIdx.x * blockDim.x + threadIdx.x;
  if (i < n) d[i] = s[i];
}
__global__ void add_kernel(float* d, const float* s, long long n) {
  long long i = (long long)blockIdx.x * blockDim.x + threadIdx.x;
  if (i < n) d[i] += s[i];
}
__global__ void init_attn_kernel(unsigned* mx, float* sums, int n) {
  int i = blockIdx.x * blockDim.x + threadIdx.x;
  if (i < n) { mx[i] = 0x007fffffu /* f2ord(-inf) */; sums[i] = 0.f; }
}

// ---------------- attention: alpha = <q[dst], k[src]+ee> / sqrt(d) ----------
__global__ void attn_alpha_kernel(const float* __restrict__ q, const float* __restrict__ k,
                                  const float* __restrict__ ee,
                                  const int* __restrict__ src, const int* __restrict__ dst,
                                  float* __restrict__ alpha, unsigned* __restrict__ mx, int nE) {
  int e = blockIdx.x * blockDim.x + threadIdx.x;
  if (e >= nE) return;
  const int s = src[e], d = dst[e];
  const float* qp = q + (size_t)d * kD;
  const float* kp = k + (size_t)s * kD;
  const float* ep = ee ? ee + (size_t)e * kD : nullptr;
  float acc = 0.f;
#pragma unroll 8
  for (int i = 0; i < kD; ++i) {
    float kj = kp[i] + (ep ? ep[i] : 0.f);
    acc += qp[i] * kj;
  }
  acc *= 0.125f;  // 1/sqrt(64)
  alpha[e] = acc;
  atomicMax(mx + d, f2ord(acc));
}

__global__ void attn_soft_kernel(float* __restrict__ alpha, const unsigned* __restrict__ mx,
                                 float* __restrict__ sums, const int* __restrict__ dst, int nE) {
  int e = blockIdx.x * blockDim.x + threadIdx.x;
  if (e >= nE) return;
  const int d = dst[e];
  float m = ord2f(mx[d]);
  if (!(m > -1e37f)) m = 0.f;  // jnp.where(isfinite(m), m, 0)
  const float p = expf(alpha[e] - m);
  alpha[e] = p;
  atomicAdd(sums + d, p);
}

__global__ void attn_agg_kernel(const float* __restrict__ v, const float* __restrict__ ee,
                                const float* __restrict__ p, const float* __restrict__ sums,
                                const int* __restrict__ src, const int* __restrict__ dst,
                                float* __restrict__ agg, int nE) {
  long long t = (long long)blockIdx.x * blockDim.x + threadIdx.x;
  if (t >= (long long)nE * kD) return;
  const int e = (int)(t >> 6), i = (int)(t & 63);
  const int s = src[e], d = dst[e];
  float vv = v[(size_t)s * kD + i] + (ee ? ee[(size_t)e * kD + i] : 0.f);
  const float a = p[e] / sums[d];
  atomicAdd(agg + (size_t)d * kD + i, a * vv);
}

// ---------------- GENConv message: agg[dst] += relu(x[src]+e) + eps ---------
__global__ void gen_msg_kernel(const float* __restrict__ xin, const float* __restrict__ e,
                               const int* __restrict__ src, const int* __restrict__ dst,
                               float* __restrict__ agg, int nE) {
  long long t = (long long)blockIdx.x * blockDim.x + threadIdx.x;
  if (t >= (long long)nE * kD) return;
  const int ed = (int)(t >> 6), i = (int)(t & 63);
  float m = xin[(size_t)src[ed] * kD + i] + e[(size_t)ed * kD + i];
  m = (m > 0.f ? m : 0.f) + kEps;
  atomicAdd(agg + (size_t)dst[ed] * kD + i, m);
}

// ---------------- augmented edge list ---------------------------------------
__global__ void build_aug_kernel(const int* __restrict__ src, const int* __restrict__ dst,
                                 const int* __restrict__ batch,
                                 int* __restrict__ as_, int* __restrict__ ad_) {
  int j = blockIdx.x * blockDim.x + threadIdx.x;
  if (j >= kEA) return;
  if (j < kE)            { as_[j] = src[j];              ad_[j] = dst[j]; }
  else if (j < kE + kN)  { int u = j - kE;      as_[j] = u;               ad_[j] = batch[u] + kN; }
  else                   { int u = j - kE - kN; as_[j] = batch[u] + kN;   ad_[j] = u; }
}

// ---------------- mean pool over segments -----------------------------------
__global__ void pool_kernel(const float* __restrict__ o, const int* __restrict__ batch,
                            float* __restrict__ glob, float* __restrict__ cnt, int nA, int N) {
  long long t = (long long)blockIdx.x * blockDim.x + threadIdx.x;
  if (t >= (long long)nA * kD) return;
  const int i = (int)(t >> 6), d = (int)(t & 63);
  const int g = (i < N) ? batch[i] : (i - N);
  atomicAdd(glob + (size_t)g * kD + d, o[(size_t)i * kD + d]);
  if (d == 0) atomicAdd(cnt + g, 1.0f);
}
__global__ void pool_div_kernel(float* glob, const float* cnt, int G) {
  long long t = (long long)blockIdx.x * blockDim.x + threadIdx.x;
  if (t >= (long long)G * kD) return;
  glob[t] /= cnt[t >> 6];
}

// ---------------- final 64 -> 1 linear --------------------------------------
__global__ void linear_n1_kernel(const float* __restrict__ A, const float* __restrict__ W,
                                 const float* __restrict__ b, float* __restrict__ out, int M) {
  int m = blockIdx.x * blockDim.x + threadIdx.x;
  if (m >= M) return;
  const float* ap = A + (size_t)m * kD;
  float acc = b[0];
#pragma unroll 8
  for (int i = 0; i < kD; ++i) acc += ap[i] * W[i];
  out[m] = acc;
}

// ---------------- gather pair + add: out[i] = o[i0[i*s]] + o[i1[i*s]] -------
__global__ void gather2_kernel(const float* __restrict__ o, const int* __restrict__ i0,
                               const int* __restrict__ i1, int stride,
                               float* __restrict__ out, int m) {
  long long t = (long long)blockIdx.x * blockDim.x + threadIdx.x;
  if (t >= (long long)m * kD) return;
  const int i = (int)(t >> 6), d = (int)(t & 63);
  out[(size_t)i * kD + d] = o[(size_t)i0[(size_t)i * stride] * kD + d] +
                            o[(size_t)i1[(size_t)i * stride] * kD + d];
}

dim3 g1(long long n) { return dim3((unsigned)((n + 255) / 256)); }

}  // namespace

extern "C" void kernel_launch(void* const* d_in, const int* in_sizes, int n_in,
                              void* d_out, int out_size, void* d_ws, size_t ws_size,
                              hipStream_t stream) {
  (void)in_sizes; (void)out_size; (void)ws_size;
  const float* x     = (const float*)d_in[0];
  const float* eattr = (const float*)d_in[1];
  const float* cond  = (const float*)d_in[2];
  const int*   eidx  = (const int*)  d_in[3];
  const int*   batch = (const int*)  d_in[4];
  const int*   neidx = (const int*)  d_in[5];
  const int* src = eidx;
  const int* dst = eidx + kE;

  // ---- parameter cursor (flat buffer at d_in[6] OR one pointer per leaf,
  //      both in pytree insertion order of the reference's params dict) ------
  struct Cursor {
    void* const* din; int idx; const float* flat; size_t off; bool flat_mode;
    const float* take(size_t n) {
      if (flat_mode) { const float* p = flat + off; off += n; return p; }
      return (const float*)din[idx++];
    }
  } cur{d_in, 6, (const float*)d_in[6], 0, n_in == 7};

  const float* x2hW = cur.take(32 * 64); const float* x2hB = cur.take(64);
  const float* e2hW = cur.take(16 * 64); const float* e2hB = cur.take(64);
  const float* c2hW = cur.take(32 * 64); const float* c2hB = cur.take(64);
  struct TCP { const float *Wq,*bq,*Wk,*bk,*Wv,*bv,*Ws,*bs,*We; };
  TCP tc[6];
  for (int i = 0; i < 6; ++i)
    tc[i] = { cur.take(4096), cur.take(64), cur.take(4096), cur.take(64),
              cur.take(4096), cur.take(64), cur.take(4096), cur.take(64), cur.take(4096) };
  struct GENP { const float *W, *b; };
  GENP gp[6];
  for (int i = 0; i < 6; ++i) gp[i] = { cur.take(4096), cur.take(64) };
  TCP oo[2];
  for (int i = 0; i < 2; ++i)
    oo[i] = { cur.take(4096), cur.take(64), cur.take(4096), cur.take(64),
              cur.take(4096), cur.take(64), cur.take(4096), cur.take(64), nullptr };
  struct HEAD { const float *W1,*b1,*W2,*b2,*W3,*b3; };
  auto takeHead = [&](int nl) {
    HEAD h; h.W1 = cur.take(4096); h.b1 = cur.take(64);
    h.W2 = cur.take(4096); h.b2 = cur.take(64);
    h.W3 = cur.take((size_t)64 * nl); h.b3 = cur.take(nl); return h;
  };
  HEAD hStop = takeHead(1), hAddN = takeHead(64), hSetN = takeHead(32),
       hAddE = takeHead(1), hSetE = takeHead(32), hRew  = takeHead(1);

  // ---- workspace carve ------------------------------------------------------
  char* base = (char*)d_ws; size_t wo = 0;
  auto carve = [&](size_t bytes) -> void* {
    void* p = base + wo; wo = (wo + bytes + 255) & ~(size_t)255; return p;
  };
  float*    o    = (float*)carve((size_t)kNA * kD * 4);
  float*    qb   = (float*)carve((size_t)kNA * kD * 4);
  float*    kb   = (float*)carve((size_t)kNA * kD * 4);
  float*    vb   = (float*)carve((size_t)kNA * kD * 4);
  float*    agg  = (float*)carve((size_t)kNA * kD * 4);
  float*    eb   = (float*)carve((size_t)kE * kD * 4);   // e embedding / head tmpB
  float*    eeb  = (float*)carve((size_t)kE * kD * 4);   // per-layer e@We / head tmpA
  float*    alph = (float*)carve((size_t)kEA * 4);
  float*    sums = (float*)carve((size_t)kNA * 4);
  unsigned* mx   = (unsigned*)carve((size_t)kNA * 4);
  int*      augs = (int*)carve((size_t)kEA * 4);
  int*      augd = (int*)carve((size_t)kEA * 4);
  float*    cb   = (float*)carve((size_t)kG * kD * 4);
  float*    glob = (float*)carve((size_t)kG * kD * 4);
  float*    cnt  = (float*)carve((size_t)kG * 4);

  // ---- input embeddings -----------------------------------------------------
  gemm_wmma<32, 64, 0><<<(kN + 127) / 128, 256, 0, stream>>>(x,     x2hW, x2hB, o,  kN);
  gemm_wmma<16, 64, 0><<<(kE + 127) / 128, 256, 0, stream>>>(eattr, e2hW, e2hB, eb, kE);
  gemm_wmma<32, 64, 0><<<(kG + 127) / 128, 256, 0, stream>>>(cond,  c2hW, c2hB, cb, kG);

  // ---- 6 x (TransformerConv + GENConv) --------------------------------------
  for (int L = 0; L < 6; ++L) {
    gemm_wmma<64, 64, 0><<<(kN + 127) / 128, 256, 0, stream>>>(o,  tc[L].Wq, tc[L].bq, qb,  kN);
    gemm_wmma<64, 64, 0><<<(kN + 127) / 128, 256, 0, stream>>>(o,  tc[L].Wk, tc[L].bk, kb,  kN);
    gemm_wmma<64, 64, 0><<<(kN + 127) / 128, 256, 0, stream>>>(o,  tc[L].Wv, tc[L].bv, vb,  kN);
    gemm_wmma<64, 64, 0><<<(kE + 127) / 128, 256, 0, stream>>>(eb, tc[L].We, nullptr,  eeb, kE);
    gemm_wmma<64, 64, 0><<<(kN + 127) / 128, 256, 0, stream>>>(o,  tc[L].Ws, tc[L].bs, agg, kN);  // skip
    init_attn_kernel<<<g1(kN), 256, 0, stream>>>(mx, sums, kN);
    attn_alpha_kernel<<<g1(kE), 256, 0, stream>>>(qb, kb, eeb, src, dst, alph, mx, kE);
    attn_soft_kernel <<<g1(kE), 256, 0, stream>>>(alph, mx, sums, dst, kE);
    attn_agg_kernel  <<<g1((long long)kE * kD), 256, 0, stream>>>(vb, eeb, alph, sums, src, dst, agg, kE);
    // GENConv: qb := agg (x + messages), then o = (qb) @ W + b
    copy_kernel<<<g1((long long)kN * kD), 256, 0, stream>>>(qb, agg, (long long)kN * kD);
    gen_msg_kernel<<<g1((long long)kE * kD), 256, 0, stream>>>(agg, eb, src, dst, qb, kE);
    gemm_wmma<64, 64, 0><<<(kN + 127) / 128, 256, 0, stream>>>(qb, gp[L].W, gp[L].b, o, kN);
  }

  // ---- augmented graph ------------------------------------------------------
  copy_kernel<<<g1((long long)kG * kD), 256, 0, stream>>>(o + (size_t)kN * kD, cb, (long long)kG * kD);
  build_aug_kernel<<<g1(kEA), 256, 0, stream>>>(src, dst, batch, augs, augd);
  for (int L = 0; L < 2; ++L) {
    gemm_wmma<64, 64, 0><<<(kNA + 127) / 128, 256, 0, stream>>>(o, oo[L].Wq, oo[L].bq, qb,  kNA);
    gemm_wmma<64, 64, 0><<<(kNA + 127) / 128, 256, 0, stream>>>(o, oo[L].Wk, oo[L].bk, kb,  kNA);
    gemm_wmma<64, 64, 0><<<(kNA + 127) / 128, 256, 0, stream>>>(o, oo[L].Wv, oo[L].bv, vb,  kNA);
    gemm_wmma<64, 64, 0><<<(kNA + 127) / 128, 256, 0, stream>>>(o, oo[L].Ws, oo[L].bs, agg, kNA);
    init_attn_kernel<<<g1(kNA), 256, 0, stream>>>(mx, sums, kNA);
    attn_alpha_kernel<<<g1(kEA), 256, 0, stream>>>(qb, kb, nullptr, augs, augd, alph, mx, kEA);
    attn_soft_kernel <<<g1(kEA), 256, 0, stream>>>(alph, mx, sums, augd, kEA);
    attn_agg_kernel  <<<g1((long long)kEA * kD), 256, 0, stream>>>(vb, nullptr, alph, sums, augs, augd, agg, kEA);
    add_kernel<<<g1((long long)kNA * kD), 256, 0, stream>>>(o, agg, (long long)kNA * kD);  // o += conv
  }

  // ---- mean pool ------------------------------------------------------------
  zero_kernel<<<g1((long long)kG * kD), 256, 0, stream>>>(glob, (long long)kG * kD);
  zero_kernel<<<g1(kG), 256, 0, stream>>>(cnt, kG);
  pool_kernel<<<g1((long long)kNA * kD), 256, 0, stream>>>(o, batch, glob, cnt, kNA, kN);
  pool_div_kernel<<<g1((long long)kG * kD), 256, 0, stream>>>(glob, cnt, kG);

  // ---- heads ----------------------------------------------------------------
  float* out   = (float*)d_out;
  float* oStop = out;
  float* oAddN = oStop + kG;
  float* oSetN = oAddN + (size_t)kN * 64;
  float* oAddE = oSetN + (size_t)kN * 32;
  float* oSetE = oAddE + (size_t)kNE;
  float* oRew  = oSetE + (size_t)kE2 * 32;
  float* tA = eeb;  // 400k x 64 scratch (e/ee no longer needed)
  float* tB = eb;

  // stop(glob)
  gemm_wmma<64, 64, 1><<<(kG + 127) / 128, 256, 0, stream>>>(glob, hStop.W1, hStop.b1, tA, kG);
  gemm_wmma<64, 64, 1><<<(kG + 127) / 128, 256, 0, stream>>>(tA,   hStop.W2, hStop.b2, tB, kG);
  linear_n1_kernel<<<g1(kG), 256, 0, stream>>>(tB, hStop.W3, hStop.b3, oStop, kG);
  // add_node(o)
  gemm_wmma<64, 64, 1><<<(kN + 127) / 128, 256, 0, stream>>>(o,  hAddN.W1, hAddN.b1, tA, kN);
  gemm_wmma<64, 64, 1><<<(kN + 127) / 128, 256, 0, stream>>>(tA, hAddN.W2, hAddN.b2, tB, kN);
  gemm_wmma<64, 64, 0><<<(kN + 127) / 128, 256, 0, stream>>>(tB, hAddN.W3, hAddN.b3, oAddN, kN);
  // set_node_attr(o)
  gemm_wmma<64, 64, 1><<<(kN + 127) / 128, 256, 0, stream>>>(o,  hSetN.W1, hSetN.b1, tA, kN);
  gemm_wmma<64, 64, 1><<<(kN + 127) / 128, 256, 0, stream>>>(tA, hSetN.W2, hSetN.b2, tB, kN);
  gemm_wmma<64, 32, 0><<<(kN + 127) / 128, 256, 0, stream>>>(tB, hSetN.W3, hSetN.b3, oSetN, kN);
  // add_edge(o[ne0] + o[ne1])
  gather2_kernel<<<g1((long long)kNE * kD), 256, 0, stream>>>(o, neidx, neidx + kNE, 1, tB, kNE);
  gemm_wmma<64, 64, 1><<<(kNE + 127) / 128, 256, 0, stream>>>(tB, hAddE.W1, hAddE.b1, tA, kNE);
  gemm_wmma<64, 64, 1><<<(kNE + 127) / 128, 256, 0, stream>>>(tA, hAddE.W2, hAddE.b2, tB, kNE);
  linear_n1_kernel<<<g1(kNE), 256, 0, stream>>>(tB, hAddE.W3, hAddE.b3, oAddE, kNE);
  // set_edge_attr(o[src[::2]] + o[dst[::2]])
  gather2_kernel<<<g1((long long)kE2 * kD), 256, 0, stream>>>(o, src, dst, 2, tB, kE2);
  gemm_wmma<64, 64, 1><<<(kE2 + 127) / 128, 256, 0, stream>>>(tB, hSetE.W1, hSetE.b1, tA, kE2);
  gemm_wmma<64, 64, 1><<<(kE2 + 127) / 128, 256, 0, stream>>>(tA, hSetE.W2, hSetE.b2, tB, kE2);
  gemm_wmma<64, 32, 0><<<(kE2 + 127) / 128, 256, 0, stream>>>(tB, hSetE.W3, hSetE.b3, oSetE, kE2);
  // reward(glob)
  gemm_wmma<64, 64, 1><<<(kG + 127) / 128, 256, 0, stream>>>(glob, hRew.W1, hRew.b1, tA, kG);
  gemm_wmma<64, 64, 1><<<(kG + 127) / 128, 256, 0, stream>>>(tA,   hRew.W2, hRew.b2, tB, kG);
  linear_n1_kernel<<<g1(kG), 256, 0, stream>>>(tB, hRew.W3, hRew.b3, oRew, kG);
}
